// RelationalGnnEncoder_49452253447021
// MI455X (gfx1250) — matrix operand order repaired and checked by reading, not compiled, not run on hardware
//
#include <hip/hip_runtime.h>
#include <hip/hip_bf16.h>

typedef __attribute__((ext_vector_type(16))) __bf16       v16bf;
typedef __attribute__((ext_vector_type(8)))  float        v8f;
typedef __attribute__((ext_vector_type(4)))  unsigned int vu4;

#define DEV __device__ __forceinline__

// ---------------------------------------------------------------------------
// WMMA fragment helpers (wave32, 16x16x32 bf16, f32 accumulate)
//
// A (16x32, row-major source): lane L holds row (L&15).
//   lanes 0-15 : elements 0..7 = K 0..7,  elements 8..15 = K 16..23
//   lanes 16-31: elements 0..7 = K 8..15, elements 8..15 = K 24..31
// B (32x16): lane L holds column (L&15); same K order as A:
//   lanes 0-15 : elements 0..7 = K 0..7, 8..15 = K 16..23 ; lanes 16-31: K 8..15, K 24..31
// C/D (16x16 f32): vgpr v, lane L -> row v + (L>=16 ? 8 : 0), col (L&15)
// ---------------------------------------------------------------------------

DEV v16bf load_a_f32(const float* __restrict__ rowptr, int lane) {
  const int cb = (lane & 16) ? 8 : 0;
  v16bf a;
#pragma unroll
  for (int i = 0; i < 8; ++i) a[i]     = (__bf16)rowptr[cb + i];
#pragma unroll
  for (int i = 0; i < 8; ++i) a[8 + i] = (__bf16)rowptr[cb + 16 + i];
  return a;
}

DEV v16bf load_a_bf16(const __bf16* __restrict__ rowptr, int lane) {
  const int cb = (lane & 16) ? 8 : 0;
  v16bf a;
#pragma unroll
  for (int i = 0; i < 8; ++i) a[i]     = rowptr[cb + i];
#pragma unroll
  for (int i = 0; i < 8; ++i) a[8 + i] = rowptr[cb + 16 + i];
  return a;
}

// Plain (non-volatile) 32B fragment read; after inlining, InferAddressSpaces
// rewrites this to LDS loads (ds_load_b128 x2). The per-iteration asm memory
// clobber in the callers stops LICM from hoisting these into live registers.
DEV v16bf lds_bfrag(const __bf16* sB, int frag, int lane) {
  return *(const v16bf*)(sB + (size_t)(frag * 32 + lane) * 16);
}

DEV v8f wmma_bf16(v16bf a, v16bf b, v8f c) {
  return __builtin_amdgcn_wmma_f32_16x16x32_bf16(
      /*neg_a=*/false, a, /*neg_b=*/false, b,
      /*c_mod=*/(short)0, c, /*reuse_a=*/false, /*reuse_b=*/false);
}

// Cooperative copy of packed fragments global -> LDS (vu4 = 16B per thread).
DEV void stage_frags(__bf16* sB, const __bf16* __restrict__ wfrag, int nfrag) {
  const vu4* gsrc = (const vu4*)wfrag;
  vu4* ldst = (vu4*)sB;
  const int nvec = nfrag * 64;  // nfrag * 1024B / 16B
  for (int i = threadIdx.x; i < nvec; i += blockDim.x) ldst[i] = gsrc[i];
  __syncthreads();
}

// ---------------------------------------------------------------------------
// Weight packing: W is f32 [Kdim x 64] row-major; produce kchunks*4 fragments
// of 32 lanes x 16 bf16 each, frag index = kc*4 + nt.
// ---------------------------------------------------------------------------
__global__ void pack_weights_kernel(const float* __restrict__ W,
                                    __bf16* __restrict__ out, int kchunks) {
  const int t = blockIdx.x * blockDim.x + threadIdx.x;
  const int nfrag = kchunks * 4;
  if (t >= nfrag * 32) return;
  const int frag = t >> 5;
  const int lane = t & 31;
  const int kc = frag >> 2;
  const int nt = frag & 3;
  const int col = nt * 16 + (lane & 15);
  const int kstart = kc * 32 + ((lane & 16) ? 8 : 0);
  __bf16* dst = out + (size_t)t * 16;
#pragma unroll
  for (int i = 0; i < 16; ++i)
    dst[i] = (__bf16)W[(size_t)(kstart + (i < 8 ? i : i + 8)) * 64 + col];
}

__global__ void zero_kernel(float* __restrict__ p, size_t n) {
  size_t i = (size_t)blockIdx.x * blockDim.x + threadIdx.x;
  const size_t stride = (size_t)gridDim.x * blockDim.x;
  for (; i < n; i += stride) p[i] = 0.f;
}

// ---------------------------------------------------------------------------
// Generic row-tiled GEMM + bias + ReLU:
//   out[rows x 64] = relu( concat(A0, A1)[rows x KCH*32] @ W + b )
// KC0 chunks come from A0 (leading dim ld0), rest from A1 (ld1).
// One wave handles a 16-row tile; grid-stride over tiles. B frags live in LDS.
// ---------------------------------------------------------------------------
template <int KCH, int KC0>
__global__ void gemm_relu_kernel(const float* __restrict__ A0, int ld0,
                                 const float* __restrict__ A1, int ld1,
                                 const __bf16* __restrict__ wfrag,
                                 const float* __restrict__ bias,
                                 float* __restrict__ out_f32,
                                 __bf16* __restrict__ out_bf16,
                                 int rows) {
  __shared__ __bf16 sB[KCH * 4 * 512];
  stage_frags(sB, wfrag, KCH * 4);

  const int lane   = threadIdx.x & 31;
  const int wave   = (blockIdx.x * blockDim.x + threadIdx.x) >> 5;
  const int nwaves = (gridDim.x * blockDim.x) >> 5;
  const int nblk   = (rows + 15) >> 4;

  const int col = lane & 15;
  float bv[4];
#pragma unroll
  for (int nt = 0; nt < 4; ++nt) bv[nt] = bias[nt * 16 + col];

  for (int blk = wave; blk < nblk; blk += nwaves) {
    asm volatile("" ::: "memory");  // block LICM of the LDS fragment loads

    const int rbase = blk << 4;
    int r = rbase + (lane & 15);
    if (r >= rows) r = rows - 1;

    v8f acc[4] = {};
#pragma unroll
    for (int kc = 0; kc < KCH; ++kc) {
      const float* src = (kc < KC0)
          ? (A0 + (size_t)r * ld0 + kc * 32)
          : (A1 + (size_t)r * ld1 + (kc - KC0) * 32);
      const v16bf a = load_a_f32(src, lane);
#pragma unroll
      for (int nt = 0; nt < 4; ++nt)
        acc[nt] = wmma_bf16(a, lds_bfrag(sB, kc * 4 + nt, lane), acc[nt]);
    }

    const int roff = (lane & 16) ? 8 : 0;
#pragma unroll
    for (int nt = 0; nt < 4; ++nt) {
#pragma unroll
      for (int v = 0; v < 8; ++v) {
        const int rr = rbase + roff + v;
        if (rr < rows) {
          float m = acc[nt][v] + bv[nt];
          m = m > 0.f ? m : 0.f;
          const size_t o = (size_t)rr * 64 + nt * 16 + col;
          if (out_f32)  out_f32[o]  = m;
          if (out_bf16) out_bf16[o] = (__bf16)m;
        }
      }
    }
  }
}

// ---------------------------------------------------------------------------
// Message + scatter kernel: per 16-edge tile
//   msg = relu( [node_state[src] | edge_state] @ W_msg + b )   (16x128 @ 128x64)
//   aggregate[tgt] += msg   via global_atomic_add_f32 (lands in 192MB L2)
// ---------------------------------------------------------------------------
__global__ void msg_scatter_kernel(const float* __restrict__ node_state,
                                   const __bf16* __restrict__ edge_state,
                                   const long long* __restrict__ eidx, // [2,E]
                                   const __bf16* __restrict__ wfrag,
                                   const float* __restrict__ bias,
                                   float* __restrict__ aggregate,
                                   int E_) {
  __shared__ __bf16 sB[16 * 512];
  stage_frags(sB, wfrag, 16);

  const int lane   = threadIdx.x & 31;
  const int wave   = (blockIdx.x * blockDim.x + threadIdx.x) >> 5;
  const int nwaves = (gridDim.x * blockDim.x) >> 5;
  const int nblk   = (E_ + 15) >> 4;

  const int col = lane & 15;
  float bv[4];
#pragma unroll
  for (int nt = 0; nt < 4; ++nt) bv[nt] = bias[nt * 16 + col];

  for (int blk = wave; blk < nblk; blk += nwaves) {
    asm volatile("" ::: "memory");  // block LICM of the LDS fragment loads

    const int ebase = blk << 4;
    int e = ebase + (lane & 15);
    if (e >= E_) e = E_ - 1;

    const int s = (int)eidx[(size_t)e];                 // source node
    const float*  nrow = node_state + (size_t)s * 64;   // gather (L2-resident)
    const __bf16* erow = edge_state + (size_t)e * 64;

    const v16bf a0 = load_a_f32(nrow, lane);
    const v16bf a1 = load_a_f32(nrow + 32, lane);
    const v16bf a2 = load_a_bf16(erow, lane);
    const v16bf a3 = load_a_bf16(erow + 32, lane);

    v8f acc[4] = {};
#pragma unroll
    for (int nt = 0; nt < 4; ++nt) {
      acc[nt] = wmma_bf16(a0, lds_bfrag(sB, 0  + nt, lane), acc[nt]);
      acc[nt] = wmma_bf16(a1, lds_bfrag(sB, 4  + nt, lane), acc[nt]);
      acc[nt] = wmma_bf16(a2, lds_bfrag(sB, 8  + nt, lane), acc[nt]);
      acc[nt] = wmma_bf16(a3, lds_bfrag(sB, 12 + nt, lane), acc[nt]);
    }

    const int roff = (lane & 16) ? 8 : 0;
#pragma unroll
    for (int v = 0; v < 8; ++v) {
      const int ee = ebase + roff + v;
      if (ee < E_) {
        const int t = (int)eidx[(size_t)E_ + ee];       // target node
        float* arow = aggregate + (size_t)t * 64;
#pragma unroll
        for (int nt = 0; nt < 4; ++nt) {
          float m = acc[nt][v] + bv[nt];
          m = m > 0.f ? m : 0.f;
          unsafeAtomicAdd(arow + nt * 16 + col, m);     // global_atomic_add_f32
        }
      }
    }
  }
}

// ---------------------------------------------------------------------------
// Host-side orchestration
// ---------------------------------------------------------------------------
extern "C" void kernel_launch(void* const* d_in, const int* in_sizes, int n_in,
                              void* d_out, int out_size, void* d_ws, size_t ws_size,
                              hipStream_t stream) {
  const float*     node_features = (const float*)d_in[0];
  const long long* edge_index    = (const long long*)d_in[1];
  const float*     edge_features = (const float*)d_in[2];
  const float*     W_node = (const float*)d_in[3];
  const float*     b_node = (const float*)d_in[4];
  const float*     W_edge = (const float*)d_in[5];
  const float*     b_edge = (const float*)d_in[6];
  const float*     W_msg  = (const float*)d_in[7];
  const float*     b_msg  = (const float*)d_in[8];
  const float*     W_upd  = (const float*)d_in[9];
  const float*     b_upd  = (const float*)d_in[10];

  const int N_ = in_sizes[0] / 128;          // D_NODE = 128
  const int E_ = in_sizes[1] / 2;            // edge_index is [2, E]
  const int L_ = in_sizes[7] / (128 * 64);   // W_msg is [L, 2H, H]

  // ---- workspace carve (256B aligned chunks) ----
  char* ws = (char*)d_ws;
  auto carve = [&](size_t bytes) -> char* {
    char* p = ws;
    ws += (bytes + 255) & ~(size_t)255;
    return p;
  };
  float*  ns0     = (float*)carve((size_t)N_ * 64 * sizeof(float));
  float*  ns1     = (float*)carve((size_t)N_ * 64 * sizeof(float));
  float*  agg     = (float*)carve((size_t)N_ * 64 * sizeof(float));
  __bf16* est     = (__bf16*)carve((size_t)E_ * 64 * sizeof(__bf16));
  __bf16* wf_node = (__bf16*)carve((size_t)16 * 512 * sizeof(__bf16));
  __bf16* wf_edge = (__bf16*)carve((size_t)8 * 512 * sizeof(__bf16));
  __bf16* wf_msg  = (__bf16*)carve((size_t)L_ * 16 * 512 * sizeof(__bf16));
  __bf16* wf_upd  = (__bf16*)carve((size_t)L_ * 16 * 512 * sizeof(__bf16));

  // ---- pack weights into WMMA B-fragment layout (tiny kernels) ----
  pack_weights_kernel<<<2, 256, 0, stream>>>(W_node, wf_node, 4);
  pack_weights_kernel<<<1, 256, 0, stream>>>(W_edge, wf_edge, 2);
  for (int l = 0; l < L_; ++l) {
    pack_weights_kernel<<<2, 256, 0, stream>>>(W_msg + (size_t)l * 128 * 64,
                                               wf_msg + (size_t)l * 16 * 512, 4);
    pack_weights_kernel<<<2, 256, 0, stream>>>(W_upd + (size_t)l * 128 * 64,
                                               wf_upd + (size_t)l * 16 * 512, 4);
  }

  const int threads = 256;                      // 8 waves/block
  const int nblkN = (N_ + 15) / 16;
  const int nblkE = (E_ + 15) / 16;
  int gridN = (nblkN + 7) / 8; if (gridN > 1024) gridN = 1024;
  int gridE = (nblkE + 7) / 8; if (gridE > 2048) gridE = 2048;

  // node encode: [N,128] @ [128,64] -> ns0 (f32)
  gemm_relu_kernel<4, 4><<<gridN, threads, 0, stream>>>(
      node_features, 128, nullptr, 0, wf_node, b_node, ns0, nullptr, N_);
  // edge encode: [E,64] @ [64,64] -> est (bf16, consumed by message layers)
  gemm_relu_kernel<2, 2><<<gridE, threads, 0, stream>>>(
      edge_features, 64, nullptr, 0, wf_edge, b_edge, nullptr, est, E_);

  float* cur = ns0;
  for (int l = 0; l < L_; ++l) {
    zero_kernel<<<256, 256, 0, stream>>>(agg, (size_t)N_ * 64);
    msg_scatter_kernel<<<gridE, threads, 0, stream>>>(
        cur, est, edge_index, wf_msg + (size_t)l * 16 * 512, b_msg + l * 64,
        agg, E_);
    float* nxt = (l == L_ - 1) ? (float*)d_out : (cur == ns0 ? ns1 : ns0);
    // update: relu([node_state | aggregate] @ W_upd + b)
    gemm_relu_kernel<4, 2><<<gridN, threads, 0, stream>>>(
        cur, 64, agg, 64, wf_upd + (size_t)l * 16 * 512, b_upd + l * 64,
        nxt, nullptr, N_);
    cur = nxt;
  }
}
